// MultiHeadAttention_33492154974615
// MI455X (gfx1250) — matrix operand order repaired and checked by reading, not compiled, not run on hardware
//
#include <hip/hip_runtime.h>

typedef __attribute__((ext_vector_type(16))) __bf16 bf16x16;
typedef __attribute__((ext_vector_type(8)))  float  f32x8;

union FragU { bf16x16 v; uint4 q[2]; };

// A-fragment (16x32, 16-bit): lane l -> row l%16, K = {klo..klo+7, klo+16..klo+23}, klo = 8*(l/16)
__device__ __forceinline__ bf16x16 load_fragA(const __bf16* p, int ld) {
  const int lane = threadIdx.x & 31;
  const int m    = lane & 15;
  const int klo  = (lane >> 4) << 3;
  const __bf16* row = p + (size_t)m * ld + klo;
  FragU f;
  f.q[0] = *(const uint4*)(row);
  f.q[1] = *(const uint4*)(row + 16);
  return f.v;
}

// B-fragment (32x16, 16-bit) loaded from row-major B^T [N][K]:
// lane l -> column l%16, K = {klo..klo+7, klo+16..klo+23}, klo = 8*(l/16)
__device__ __forceinline__ bf16x16 load_fragB(const __bf16* p, int ld) {
  const int lane = threadIdx.x & 31;
  const int n    = lane & 15;
  const int ko   = (lane >> 4) << 3;
  const __bf16* row = p + (size_t)n * ld + ko;
  FragU f;
  f.q[0] = *(const uint4*)(row);
  f.q[1] = *(const uint4*)(row + 16);
  return f.v;
}

__device__ __forceinline__ f32x8 wmma_bf16(bf16x16 a, bf16x16 b, f32x8 c) {
  return __builtin_amdgcn_wmma_f32_16x16x32_bf16(false, a, false, b, (short)0, c, false, false);
}

// CDNA5 async global->LDS copy, SADDR form: scalar 64-bit base + per-lane 32-bit byte offset.
// Keeps per-iteration address math on the SALU; transfer is memory->LDS, ASYNCcnt-tracked.
__device__ __forceinline__ void async_g2l_b128(unsigned lds_off, unsigned voff,
                                               unsigned long long sbase) {
  asm volatile("global_load_async_to_lds_b128 %0, %1, %2"
               :: "v"(lds_off), "v"(voff), "s"(sbase)
               : "memory");
}
__device__ __forceinline__ void wait_async0() {
  asm volatile("s_wait_asynccnt 0" ::: "memory");
}

// ---------------- conversion / transpose helpers ----------------

__global__ void f32_to_bf16_kernel(const float* __restrict__ in, __bf16* __restrict__ out, int n4) {
  int i = blockIdx.x * 256 + threadIdx.x; // one float4 per thread
  if (i < n4) {
    float4 v = ((const float4*)in)[i];
    __bf16* o = out + (size_t)i * 4;
    o[0] = (__bf16)v.x; o[1] = (__bf16)v.y; o[2] = (__bf16)v.z; o[3] = (__bf16)v.w;
  }
}

// W [K][N] f32 -> Wt [N][K] bf16
__global__ void transpose_bf16_kernel(const float* __restrict__ W, __bf16* __restrict__ Wt,
                                      int K, int N) {
  int n = blockIdx.x * 32 + (threadIdx.x & 31);
  int k = blockIdx.y * 8 + (threadIdx.x >> 5);
  if (n < N && k < K) Wt[(size_t)n * K + k] = (__bf16)W[(size_t)k * N + n];
}

// ------- shared GEMM mainloop: C[128x64] += A[128x1024] * Bt^T, async double-buffered -------

#define AS_ELEMS (128 * 64)
#define BS_ELEMS (64 * 64)

struct IssueCtx {
  unsigned voffA[4], ldsA[4];  // per-thread, loop-invariant
  unsigned voffB[2], ldsB[2];
};

__device__ __forceinline__ void gemm_issue_tile(const IssueCtx& cx,
                                                unsigned long long aBase, unsigned long long bBase,
                                                int k0, int buf) {
  // fold the uniform K advance into the scalar base (bytes)
  const unsigned long long aB = aBase + (unsigned long long)k0 * 2u;
  const unsigned long long bB = bBase + (unsigned long long)k0 * 2u;
  const unsigned aSel = (unsigned)buf * (AS_ELEMS * 2u);
  const unsigned bSel = (unsigned)buf * (BS_ELEMS * 2u);
  #pragma unroll
  for (int j = 0; j < 4; ++j)              // A tile: 128x64 bf16 = 1024 x 16B
    async_g2l_b128(cx.ldsA[j] + aSel, cx.voffA[j], aB);
  #pragma unroll
  for (int j = 0; j < 2; ++j)              // B tile: 64x64 bf16 = 512 x 16B
    async_g2l_b128(cx.ldsB[j] + bSel, cx.voffB[j], bB);
}

__device__ __forceinline__ void gemm_tile(const __bf16* __restrict__ A,
                                          const __bf16* __restrict__ Bt,
                                          __bf16* As, __bf16* Bs, // 2 buffers each
                                          int m0, int n0, f32x8 acc[2][2]) {
  const int t = threadIdx.x;               // 256 threads
  const int wave = t >> 5;
  const int wm = (wave >> 1) * 32;
  const int wn = (wave & 1) * 32;
  const unsigned asBase = (unsigned)(unsigned long long)(size_t)As; // LDS byte offset
  const unsigned bsBase = (unsigned)(unsigned long long)(size_t)Bs;
  const unsigned long long aBase = (unsigned long long)(size_t)A;
  const unsigned long long bBase = (unsigned long long)(size_t)Bt;

  IssueCtx cx;
  #pragma unroll
  for (int j = 0; j < 4; ++j) {
    int li = t + 256 * j;
    int r = li >> 3, c = li & 7;
    cx.voffA[j] = (unsigned)((m0 + r) * 2048 + c * 16);  // ldA = 1024 bf16 = 2048 B
    cx.ldsA[j]  = asBase + (unsigned)li * 16u;
  }
  #pragma unroll
  for (int j = 0; j < 2; ++j) {
    int li = t + 256 * j;
    int r = li >> 3, c = li & 7;
    cx.voffB[j] = (unsigned)((n0 + r) * 2048 + c * 16);  // ldB = 1024 bf16 = 2048 B
    cx.ldsB[j]  = bsBase + (unsigned)li * 16u;
  }

  gemm_issue_tile(cx, aBase, bBase, 0, 0);  // prologue: buffer 0
  int cur = 0;
  for (int k0 = 0; k0 < 1024; k0 += 64) {
    wait_async0();       // this wave's async copies into buf[cur] are done
    __syncthreads();     // everyone's are, and prior readers of buf[cur^1] finished
    if (k0 + 64 < 1024)  // overlap: prefetch next K-tile into the other buffer
      gemm_issue_tile(cx, aBase, bBase, k0 + 64, cur ^ 1);
    const __bf16* Ac = As + cur * AS_ELEMS;
    const __bf16* Bc = Bs + cur * BS_ELEMS;
    #pragma unroll
    for (int ks = 0; ks < 64; ks += 32) {
      bf16x16 a0 = load_fragA(Ac + (size_t)(wm + 0) * 64 + ks, 64);
      bf16x16 a1 = load_fragA(Ac + (size_t)(wm + 16) * 64 + ks, 64);
      bf16x16 b0 = load_fragB(Bc + (size_t)(wn + 0) * 64 + ks, 64);
      bf16x16 b1 = load_fragB(Bc + (size_t)(wn + 16) * 64 + ks, 64);
      acc[0][0] = wmma_bf16(a0, b0, acc[0][0]);
      acc[0][1] = wmma_bf16(a0, b1, acc[0][1]);
      acc[1][0] = wmma_bf16(a1, b0, acc[1][0]);
      acc[1][1] = wmma_bf16(a1, b1, acc[1][1]);
    }
    cur ^= 1;
  }
}

// ---------------- QKV GEMM: writes Q,K [bh][s][64] (Q pre-scaled), V transposed [bh][64][s]

__global__ __launch_bounds__(256) void gemm_qkv_kernel(
    const __bf16* __restrict__ A, const __bf16* __restrict__ Bt, const float* __restrict__ bias,
    __bf16* __restrict__ qb, __bf16* __restrict__ kb, __bf16* __restrict__ vt) {
  __shared__ __align__(16) __bf16 As[2 * AS_ELEMS];
  __shared__ __align__(16) __bf16 Bs[2 * BS_ELEMS];
  const int m0 = blockIdx.y * 128, n0 = blockIdx.x * 64;
  f32x8 acc[2][2] = {};
  gemm_tile(A, Bt, As, Bs, m0, n0, acc);

  const int lane = threadIdx.x & 31, wave = threadIdx.x >> 5;
  const int wm = (wave >> 1) * 32, wn = (wave & 1) * 32;
  const int half = lane >> 4, cn = lane & 15;
  #pragma unroll
  for (int i = 0; i < 2; ++i)
    #pragma unroll
    for (int j = 0; j < 2; ++j) {
      const int colb = n0 + wn + 16 * j + cn;
      const float bv = bias[colb];
      const int which = colb >> 10, h = (colb >> 6) & 15, d = colb & 63;
      #pragma unroll
      for (int r = 0; r < 8; ++r) {
        const int row = m0 + wm + 16 * i + r + 8 * half;
        const int b = row >> 11, s = row & 2047;
        const float v = acc[i][j][r] + bv;
        const size_t bh = (size_t)(b * 16 + h);
        if (which == 0)      qb[(bh * 2048 + s) * 64 + d] = (__bf16)(v * 0.125f); // fold 1/sqrt(64)
        else if (which == 1) kb[(bh * 2048 + s) * 64 + d] = (__bf16)v;
        else                 vt[(bh * 64 + d) * 2048 + s] = (__bf16)v;
      }
    }
}

// ---------------- Flash attention: one wave = 16 query rows, stream 32 keys/iter ----

__global__ __launch_bounds__(128) void attn_kernel(
    const __bf16* __restrict__ qb, const __bf16* __restrict__ kb,
    const __bf16* __restrict__ vt, __bf16* __restrict__ ao) {
  __shared__ __align__(16) __bf16 Ps[4][16 * 32]; // per-wave P staging
  const int wave = threadIdx.x >> 5, lane = threadIdx.x & 31;
  const int half = lane >> 4, cn = lane & 15;
  const int bh = blockIdx.y, b = bh >> 4, h = bh & 15;
  const int s0 = blockIdx.x * 64 + wave * 16;

  const __bf16* qp = qb + ((size_t)bh * 2048 + s0) * 64;
  bf16x16 aQ0 = load_fragA(qp, 64);        // Hd 0..31
  bf16x16 aQ1 = load_fragA(qp + 32, 64);   // Hd 32..63

  f32x8 oacc[4] = {};
  float mi[8], li[8];
  #pragma unroll
  for (int r = 0; r < 8; ++r) { mi[r] = -3.0e38f; li[r] = 0.f; }

  for (int j0 = 0; j0 < 2048; j0 += 32) {
    const __bf16* kp = kb + ((size_t)bh * 2048 + j0) * 64;
    f32x8 st[2];
    #pragma unroll
    for (int tg = 0; tg < 2; ++tg) {       // keys j0+16*tg .. +15
      bf16x16 bK0 = load_fragB(kp + (size_t)(tg * 16) * 64, 64);
      bf16x16 bK1 = load_fragB(kp + (size_t)(tg * 16) * 64 + 32, 64);
      f32x8 z = {};
      z = wmma_bf16(aQ0, bK0, z);
      z = wmma_bf16(aQ1, bK1, z);
      st[tg] = z;
    }
    // online softmax (row r lives in one 16-lane half; shfl_xor masks <=8 stay in-half)
    float mnew[8], alpha[8];
    #pragma unroll
    for (int r = 0; r < 8; ++r) {
      float v = fmaxf(st[0][r], st[1][r]);
      v = fmaxf(v, __shfl_xor(v, 1));
      v = fmaxf(v, __shfl_xor(v, 2));
      v = fmaxf(v, __shfl_xor(v, 4));
      v = fmaxf(v, __shfl_xor(v, 8));
      mnew[r]  = fmaxf(mi[r], v);
      alpha[r] = __expf(mi[r] - mnew[r]);
      mi[r]    = mnew[r];
    }
    #pragma unroll
    for (int tg = 0; tg < 2; ++tg)
      #pragma unroll
      for (int r = 0; r < 8; ++r)
        st[tg][r] = __expf(st[tg][r] - mnew[r]);
    #pragma unroll
    for (int r = 0; r < 8; ++r) {
      float v = st[0][r] + st[1][r];
      v += __shfl_xor(v, 1);
      v += __shfl_xor(v, 2);
      v += __shfl_xor(v, 4);
      v += __shfl_xor(v, 8);
      li[r] = li[r] * alpha[r] + v;
    }
    #pragma unroll
    for (int nt = 0; nt < 4; ++nt)
      #pragma unroll
      for (int r = 0; r < 8; ++r)
        oacc[nt][r] *= alpha[r];
    // D-layout -> A-fragment transpose of P through per-wave LDS
    #pragma unroll
    for (int tg = 0; tg < 2; ++tg)
      #pragma unroll
      for (int r = 0; r < 8; ++r)
        Ps[wave][(r + 8 * half) * 32 + tg * 16 + cn] = (__bf16)st[tg][r];
    asm volatile("s_wait_dscnt 0" ::: "memory");
    bf16x16 aP = load_fragA(&Ps[wave][0], 32);
    #pragma unroll
    for (int nt = 0; nt < 4; ++nt) {
      bf16x16 bV = load_fragB(vt + ((size_t)bh * 64 + nt * 16) * 2048 + j0, 2048);
      oacc[nt] = wmma_bf16(aP, bV, oacc[nt]);
    }
  }
  // normalize & store bf16 [b][s][1024]
  #pragma unroll
  for (int r = 0; r < 8; ++r) {
    const float inv = 1.0f / li[r];
    const int s = s0 + r + 8 * half;
    #pragma unroll
    for (int nt = 0; nt < 4; ++nt)
      ao[(size_t)(b * 2048 + s) * 1024 + h * 64 + nt * 16 + cn] = (__bf16)(oacc[nt][r] * inv);
  }
}

// ---------------- output projection -> fp32 ----------------

__global__ __launch_bounds__(256) void gemm_proj_kernel(
    const __bf16* __restrict__ A, const __bf16* __restrict__ Bt,
    const float* __restrict__ bias, float* __restrict__ C) {
  __shared__ __align__(16) __bf16 As[2 * AS_ELEMS];
  __shared__ __align__(16) __bf16 Bs[2 * BS_ELEMS];
  const int m0 = blockIdx.y * 128, n0 = blockIdx.x * 64;
  f32x8 acc[2][2] = {};
  gemm_tile(A, Bt, As, Bs, m0, n0, acc);

  const int lane = threadIdx.x & 31, wave = threadIdx.x >> 5;
  const int wm = (wave >> 1) * 32, wn = (wave & 1) * 32;
  const int half = lane >> 4, cn = lane & 15;
  #pragma unroll
  for (int i = 0; i < 2; ++i)
    #pragma unroll
    for (int j = 0; j < 2; ++j) {
      const int colb = n0 + wn + 16 * j + cn;
      const float bv = bias[colb];
      #pragma unroll
      for (int r = 0; r < 8; ++r) {
        const int row = m0 + wm + 16 * i + r + 8 * half;
        C[(size_t)row * 1024 + colb] = acc[i][j][r] + bv;
      }
    }
}

// ---------------- launch ----------------

extern "C" void kernel_launch(void* const* d_in, const int* in_sizes, int n_in,
                              void* d_out, int out_size, void* d_ws, size_t ws_size,
                              hipStream_t stream) {
  (void)in_sizes; (void)n_in; (void)out_size; (void)ws_size;
  const float* x     = (const float*)d_in[0];
  const float* Wqkv  = (const float*)d_in[1];
  const float* bqkv  = (const float*)d_in[2];
  const float* Wproj = (const float*)d_in[3];
  const float* bproj = (const float*)d_in[4];
  float* out = (float*)d_out;

  char* ws = (char*)d_ws;
  __bf16* xbf    = (__bf16*)(ws);                   // 8 MB  : x bf16 [4096][1024]
  __bf16* wqkvT  = (__bf16*)(ws + (8ull  << 20));   // 6 MB  : W_qkv^T [3072][1024]
  __bf16* wprojT = (__bf16*)(ws + (14ull << 20));   // 2 MB  : W_proj^T [1024][1024]
  __bf16* qb     = (__bf16*)(ws + (16ull << 20));   // 8 MB  : Q [32][2048][64] (pre-scaled)
  __bf16* kbuf   = (__bf16*)(ws + (24ull << 20));   // 8 MB  : K [32][2048][64]
  __bf16* vt     = (__bf16*)(ws + (32ull << 20));   // 8 MB  : V^T [32][64][2048]
  __bf16* ao     = (__bf16*)(ws + (40ull << 20));   // 8 MB  : attn out bf16 [4096][1024]

  f32_to_bf16_kernel<<<dim3((4096 * 1024 / 4) / 256), 256, 0, stream>>>(x, xbf, 4096 * 1024 / 4);
  transpose_bf16_kernel<<<dim3(3072 / 32, 1024 / 8), 256, 0, stream>>>(Wqkv, wqkvT, 1024, 3072);
  transpose_bf16_kernel<<<dim3(1024 / 32, 1024 / 8), 256, 0, stream>>>(Wproj, wprojT, 1024, 1024);
  gemm_qkv_kernel<<<dim3(3072 / 64, 4096 / 128), 256, 0, stream>>>(xbf, wqkvT, bqkv, qb, kbuf, vt);
  attn_kernel<<<dim3(2048 / 64, 32), 128, 0, stream>>>(qb, kbuf, vt, ao);
  gemm_proj_kernel<<<dim3(1024 / 64, 4096 / 128), 256, 0, stream>>>(ao, wprojT, bproj, out);
}
